// AdaptiveQuadratureHead_60464549593748
// MI455X (gfx1250) — matrix operand order repaired and checked by reading, not compiled, not run on hardware
//
#include <hip/hip_runtime.h>
#include <math.h>

typedef __attribute__((ext_vector_type(16))) _Float16 v16h;
typedef __attribute__((ext_vector_type(8)))  float    v8f;

// ---- problem constants ----
#define B_    8
#define N_    8192
#define BN_   65536          // B_*N_
#define DXUP_ 96             // 68 padded to 96 (3 K-chunks of 32)
#define P_    128
#define DOUT_ 128
#define NSPLIT_ 16

// =====================================================================
// WMMA helpers (CDNA5 v_wmma_f32_16x16x32_f16)
// =====================================================================
__device__ __forceinline__ v8f wmma_f(v16h a, v16h b, v8f c) {
    return __builtin_amdgcn_wmma_f32_16x16x32_f16(
        /*neg_a=*/false, a, /*neg_b=*/false, b,
        /*c_mod=*/(short)0, c, /*reuse_a=*/false, /*reuse_b=*/false);
}

// A fragment (16xK tile, row-major source, ld in halves). Lane l: row = l&15,
// half = l>>4. Two contiguous 16B runs: [kc+half*8, +8) and [kc+16+half*8, +8).
__device__ __forceinline__ v16h load_afrag(const _Float16* base, int ld, int kc, int lane) {
    int m = lane & 15, hf = lane >> 4;
    const _Float16* p = base + (size_t)m * ld + kc + hf * 8;
    union { v16h h; uint4 q[2]; } f;
    f.q[0] = *(const uint4*)p;
    f.q[1] = *(const uint4*)(p + 16);
    return f.h;
}

// B fragment from "transposed" storage: element (k, col) lives at wt[col*ld + k].
// Lane l: col = colbase + (l&15), khalf = l>>4; two 16B runs: [kc+khalf*8, +8) and [kc+16+khalf*8, +8).
__device__ __forceinline__ v16h load_bfrag(const _Float16* wt, size_t ld, int colbase, int kc, int lane) {
    int col = colbase + (lane & 15), kh = lane >> 4;
    const _Float16* p = wt + (size_t)col * ld + kc + kh * 8;
    union { v16h h; uint4 q[2]; } f;
    f.q[0] = *(const uint4*)p;
    f.q[1] = *(const uint4*)(p + 16);
    return f.h;
}

__device__ __forceinline__ float gelu_f(float x) {
    return 0.5f * x * (1.0f + erff(x * 0.70710678118654752f));
}

// =====================================================================
// conversion / init kernels
// =====================================================================
__global__ void conv_weight_kernel(const float* __restrict__ src, _Float16* __restrict__ dst,
                                   int Kin, int Kpad, int O) {
    int i = blockIdx.x * 256 + threadIdx.x;
    if (i >= O * Kpad) return;
    int o = i / Kpad, k = i % Kpad;
    dst[i] = (k < Kin) ? (_Float16)src[(size_t)k * O + o] : (_Float16)0.0f;
}

__global__ void conv_x16_kernel(const float* __restrict__ x, _Float16* __restrict__ x16) {
    size_t i = (size_t)blockIdx.x * 256 + threadIdx.x;   // BN_*64 threads
    x16[i] = (_Float16)x[i];
}

__global__ void conv_xu_kernel(const float* __restrict__ x, const float* __restrict__ u,
                               _Float16* __restrict__ xu16) {
    size_t i = (size_t)blockIdx.x * 256 + threadIdx.x;   // BN_*96 threads
    size_t row = i / DXUP_;
    int c = (int)(i % DXUP_);
    float v = (c < 64) ? x[row * 64 + c] : ((c < 68) ? u[row * 4 + (c - 64)] : 0.0f);
    xu16[i] = (_Float16)v;
}

__global__ void conv_w_kernel(const int* __restrict__ mask, const float* __restrict__ sw,
                              float* __restrict__ wq) {
    size_t i = (size_t)blockIdx.x * 256 + threadIdx.x;   // BN_ threads
    wq[i] = fmaxf(sw[i], 0.0f) * (mask[i] != 0 ? 1.0f : 0.0f);
}

__global__ void zero_kernel(float* __restrict__ p) {
    size_t i = (size_t)blockIdx.x * 256 + threadIdx.x;
    p[i] = 0.0f;
}

// =====================================================================
// fused 3-layer MLP: in(BN x Kpad,f16) -> gelu -> 256 -> gelu -> 64
// one wave per 16-row tile; N-tiles processed in groups of 4 so only
// 4 accumulator fragments (32 VGPRs) are live at once -> no spills.
// LDS double buffer: h1 (layer1 out) and h2 (layer2 out).
// vmode=0: out row-major (BN x 64); vmode=1: out transposed (B,64,N)
// =====================================================================
__global__ void __launch_bounds__(128, 1)
mlp3_kernel(const _Float16* __restrict__ in, int ldin, int nk1,
            const _Float16* __restrict__ w1t, const float* __restrict__ b1,
            const _Float16* __restrict__ w2t, const float* __restrict__ b2,
            const _Float16* __restrict__ w3t, const float* __restrict__ b3,
            _Float16* __restrict__ out, int vmode) {
    __shared__ _Float16 lds[4][2][16 * 256];   // 64 KB / block
    int wave = threadIdx.x >> 5, lane = threadIdx.x & 31;
    _Float16* h1 = lds[wave][0];
    _Float16* h2 = lds[wave][1];
    size_t tile  = (size_t)blockIdx.x * 4 + wave;
    size_t rbase = tile * 16;
    const _Float16* inrow = in + rbase * ldin;
    int col = lane & 15, r0 = (lane >> 4) * 8;

    __builtin_prefetch(w2t, 0, 0);

    v8f zero = {};
    // ---- layer 1: (16 x Kpad) @ (Kpad x 256), 4 tile-groups of 4 ----
    for (int g = 0; g < 4; g++) {
        v8f c[4];
#pragma unroll
        for (int t = 0; t < 4; t++) c[t] = zero;
        for (int kc = 0; kc < nk1 * 32; kc += 32) {
            v16h a = load_afrag(inrow, ldin, kc, lane);
#pragma unroll
            for (int t = 0; t < 4; t++)
                c[t] = wmma_f(a, load_bfrag(w1t, ldin, (g * 4 + t) * 16, kc, lane), c[t]);
        }
#pragma unroll
        for (int t = 0; t < 4; t++) {
            int cb = (g * 4 + t) * 16 + col;
            float bv = b1[cb];
#pragma unroll
            for (int v = 0; v < 8; v++)
                h1[(r0 + v) * 256 + cb] = (_Float16)gelu_f(c[t][v] + bv);
        }
    }
    asm volatile("s_wait_dscnt 0" ::: "memory");

    // ---- layer 2: (16 x 256) @ (256 x 256), read h1, write h2 ----
    for (int g = 0; g < 4; g++) {
        v8f c[4];
#pragma unroll
        for (int t = 0; t < 4; t++) c[t] = zero;
        for (int kc = 0; kc < 256; kc += 32) {
            v16h a = load_afrag(h1, 256, kc, lane);
#pragma unroll
            for (int t = 0; t < 4; t++)
                c[t] = wmma_f(a, load_bfrag(w2t, 256, (g * 4 + t) * 16, kc, lane), c[t]);
        }
#pragma unroll
        for (int t = 0; t < 4; t++) {
            int cb = (g * 4 + t) * 16 + col;
            float bv = b2[cb];
#pragma unroll
            for (int v = 0; v < 8; v++)
                h2[(r0 + v) * 256 + cb] = (_Float16)gelu_f(c[t][v] + bv);
        }
    }
    asm volatile("s_wait_dscnt 0" ::: "memory");

    // ---- layer 3: (16 x 256) @ (256 x 64), bias only ----
    v8f c3[4];
#pragma unroll
    for (int t = 0; t < 4; t++) c3[t] = zero;
    for (int kc = 0; kc < 256; kc += 32) {
        v16h a = load_afrag(h2, 256, kc, lane);
#pragma unroll
        for (int t = 0; t < 4; t++)
            c3[t] = wmma_f(a, load_bfrag(w3t, 256, t * 16, kc, lane), c3[t]);
    }
    if (!vmode) {
#pragma unroll
        for (int t = 0; t < 4; t++) {
            float bv = b3[t * 16 + col];
#pragma unroll
            for (int v = 0; v < 8; v++)
                out[(rbase + r0 + v) * 64 + t * 16 + col] = (_Float16)(c3[t][v] + bv);
        }
    } else {
        size_t bidx = rbase / N_;
        size_t nloc = rbase % N_;
#pragma unroll
        for (int t = 0; t < 4; t++) {
            float bv = b3[t * 16 + col];
#pragma unroll
            for (int v = 0; v < 8; v++)
                out[((bidx * 64) + t * 16 + col) * N_ + nloc + r0 + v] = (_Float16)(c3[t][v] + bv);
        }
    }
}

// =====================================================================
// denom + context vector:  denom[b]=max(sum w,eps); c[b,d]=sum(w*V)/denom
// =====================================================================
__global__ void ctx_kernel(const float* __restrict__ wq, const _Float16* __restrict__ v16t,
                           float* __restrict__ denom, float* __restrict__ cvec) {
    int b = blockIdx.x, tid = threadIdx.x;
    __shared__ float red[256];
    float s = 0.0f;
    for (int n = tid; n < N_; n += 256) s += wq[(size_t)b * N_ + n];
    red[tid] = s;
    __syncthreads();
    for (int st = 128; st > 0; st >>= 1) {
        if (tid < st) red[tid] += red[tid + st];
        __syncthreads();
    }
    float dn = fmaxf(red[0], 1e-8f);
    if (tid == 0) denom[b] = dn;
    __syncthreads();

    int d = tid & 63, g = tid >> 6;            // 4 partial sums per d
    float cs = 0.0f;
    const _Float16* vp = v16t + ((size_t)b * 64 + d) * N_;
    const float* wp = wq + (size_t)b * N_;
    for (int n = g * 2048; n < (g + 1) * 2048; ++n) cs += wp[n] * (float)vp[n];
    red[tid] = cs;
    __syncthreads();
    if (g == 0) cvec[b * 64 + d] = (red[d] + red[d + 64] + red[d + 128] + red[d + 192]) / dn;
}

// =====================================================================
// adaptive query branch (tiny): c -> gelu -> 512 -> tanh*0.1 -> Q (f16)
// =====================================================================
__global__ void adapt_kernel(const float* __restrict__ cvec,
                             const float* __restrict__ aW1, const float* __restrict__ ab1,
                             const float* __restrict__ aW2, const float* __restrict__ ab2,
                             const float* __restrict__ query, const float* __restrict__ Bdirs,
                             _Float16* __restrict__ q16) {
    int b = blockIdx.x, tid = threadIdx.x;   // 128 threads
    __shared__ float cs[64], h1[64], av[512];
    if (tid < 64) cs[tid] = cvec[b * 64 + tid];
    __syncthreads();
    if (tid < 64) {
        float s = ab1[tid];
        for (int k = 0; k < 64; k++) s += cs[k] * aW1[k * 64 + tid];
        h1[tid] = gelu_f(s);
    }
    __syncthreads();
    for (int j = 0; j < 4; j++) {
        int o = tid * 4 + j;
        float s = ab2[o];
        for (int k = 0; k < 64; k++) s += h1[k] * aW2[k * 512 + o];
        av[o] = tanhf(s) * 0.1f;
    }
    __syncthreads();
    float a0 = av[tid * 4], a1 = av[tid * 4 + 1], a2 = av[tid * 4 + 2], a3 = av[tid * 4 + 3];
    for (int k = 0; k < 64; k++) {
        float q = query[tid * 64 + k] + a0 * Bdirs[k] + a1 * Bdirs[64 + k]
                + a2 * Bdirs[128 + k] + a3 * Bdirs[192 + k];
        q16[((size_t)b * P_ + tid) * 64 + k] = (_Float16)q;
    }
}

// =====================================================================
// fused attention: scores (WMMA) -> softplus*w -> LDS relayout -> pooled (WMMA)
// one wave per (b, p-tile, n-split); partial sums via f32 atomics
// =====================================================================
__global__ void __launch_bounds__(128, 1)
attn_kernel(const _Float16* __restrict__ q16, const _Float16* __restrict__ k16,
            const _Float16* __restrict__ v16t, const float* __restrict__ wq,
            float* __restrict__ pooled) {
    __shared__ _Float16 lds[4][16 * 32];
    int wave = threadIdx.x >> 5, lane = threadIdx.x & 31;
    _Float16* ph = lds[wave];
    int wid = blockIdx.x * 4 + wave;
    int ns = wid & (NSPLIT_ - 1), pt = (wid >> 4) & 7, b = wid >> 7;
    int col = lane & 15, r0 = (lane >> 4) * 8;

    const _Float16* qbase = q16 + ((size_t)b * P_ + pt * 16) * 64;
    v16h aq0 = load_afrag(qbase, 64, 0, lane);
    v16h aq1 = load_afrag(qbase, 64, 32, lane);

    v8f zero = {};
    v8f pc0 = zero, pc1 = zero, pc2 = zero, pc3 = zero;
    const int span = N_ / NSPLIT_;             // 512 sensors per wave

    for (int n0 = ns * span; n0 < ns * span + span; n0 += 32) {
#pragma unroll
        for (int s = 0; s < 2; s++) {
            const _Float16* kb = k16 + ((size_t)b * N_ + n0 + s * 16) * 64;
            v8f sd = zero;
            sd = wmma_f(aq0, load_bfrag(kb, 64, 0, 0, lane), sd);
            sd = wmma_f(aq1, load_bfrag(kb, 64, 0, 32, lane), sd);
            float wv = wq[(size_t)b * N_ + n0 + s * 16 + col];
#pragma unroll
            for (int v = 0; v < 8; v++) {
                float x = sd[v] * 0.125f;                  // / sqrt(DK)
                float sp = (x > 20.0f) ? x : log1pf(expf(x));
                ph[(r0 + v) * 32 + s * 16 + col] = (_Float16)(sp * wv);
            }
        }
        asm volatile("s_wait_dscnt 0" ::: "memory");
        v16h pa = load_afrag(ph, 32, 0, lane);
        const _Float16* vb = v16t + (size_t)b * 64 * N_;
        pc0 = wmma_f(pa, load_bfrag(vb, N_, 0,  n0, lane), pc0);
        pc1 = wmma_f(pa, load_bfrag(vb, N_, 16, n0, lane), pc1);
        pc2 = wmma_f(pa, load_bfrag(vb, N_, 32, n0, lane), pc2);
        pc3 = wmma_f(pa, load_bfrag(vb, N_, 48, n0, lane), pc3);
        asm volatile("s_wait_dscnt 0" ::: "memory");
    }
    float* pr = pooled + ((size_t)b * P_ + pt * 16) * 64;
#pragma unroll
    for (int v = 0; v < 8; v++) {
        atomicAdd(pr + (r0 + v) * 64 + col +  0, pc0[v]);
        atomicAdd(pr + (r0 + v) * 64 + col + 16, pc1[v]);
        atomicAdd(pr + (r0 + v) * 64 + col + 32, pc2[v]);
        atomicAdd(pr + (r0 + v) * 64 + col + 48, pc3[v]);
    }
}

__global__ void pooled_norm_kernel(const float* __restrict__ pooled,
                                   const float* __restrict__ denom,
                                   _Float16* __restrict__ p16) {
    size_t i = (size_t)blockIdx.x * 256 + threadIdx.x;   // 65536
    int b = (int)(i >> 13);                              // / (128*64)
    p16[i] = (_Float16)(pooled[i] / denom[b]);
}

// =====================================================================
// rho MLP: pooled16(1024x64) -> gelu(256) -> 128, f32 out
// =====================================================================
__global__ void __launch_bounds__(128, 1)
mlp_rho_kernel(const _Float16* __restrict__ in,
               const _Float16* __restrict__ w1t, const float* __restrict__ b1,
               const _Float16* __restrict__ w2t, const float* __restrict__ b2,
               float* __restrict__ out) {
    __shared__ _Float16 lds[4][16 * 256];   // 32 KB / block
    int wave = threadIdx.x >> 5, lane = threadIdx.x & 31;
    _Float16* h = lds[wave];
    size_t tile  = (size_t)blockIdx.x * 4 + wave;
    size_t rbase = tile * 16;
    const _Float16* inrow = in + rbase * 64;
    int col = lane & 15, r0 = (lane >> 4) * 8;

    v8f zero = {};
    // layer 1: (16x64)@(64x256), 4 groups of 4 tiles
    for (int g = 0; g < 4; g++) {
        v8f c[4];
#pragma unroll
        for (int t = 0; t < 4; t++) c[t] = zero;
        for (int kc = 0; kc < 64; kc += 32) {
            v16h a = load_afrag(inrow, 64, kc, lane);
#pragma unroll
            for (int t = 0; t < 4; t++)
                c[t] = wmma_f(a, load_bfrag(w1t, 64, (g * 4 + t) * 16, kc, lane), c[t]);
        }
#pragma unroll
        for (int t = 0; t < 4; t++) {
            int cb = (g * 4 + t) * 16 + col;
            float bv = b1[cb];
#pragma unroll
            for (int v = 0; v < 8; v++)
                h[(r0 + v) * 256 + cb] = (_Float16)gelu_f(c[t][v] + bv);
        }
    }
    asm volatile("s_wait_dscnt 0" ::: "memory");

    // layer 2: (16x256)@(256x128), 2 groups of 4 tiles, f32 out
    for (int g = 0; g < 2; g++) {
        v8f c[4];
#pragma unroll
        for (int t = 0; t < 4; t++) c[t] = zero;
        for (int kc = 0; kc < 256; kc += 32) {
            v16h a = load_afrag(h, 256, kc, lane);
#pragma unroll
            for (int t = 0; t < 4; t++)
                c[t] = wmma_f(a, load_bfrag(w2t, 256, (g * 4 + t) * 16, kc, lane), c[t]);
        }
#pragma unroll
        for (int t = 0; t < 4; t++) {
            int cb = (g * 4 + t) * 16 + col;
            float bv = b2[cb];
#pragma unroll
            for (int v = 0; v < 8; v++)
                out[(rbase + r0 + v) * DOUT_ + cb] = c[t][v] + bv;
        }
    }
}

// =====================================================================
// host launcher
// =====================================================================
extern "C" void kernel_launch(void* const* d_in, const int* in_sizes, int n_in,
                              void* d_out, int out_size, void* d_ws, size_t ws_size,
                              hipStream_t stream) {
    (void)in_sizes; (void)n_in; (void)out_size; (void)ws_size;
    char* ws = (char*)d_ws;
    size_t o = 0;
    auto take = [&](size_t bytes) { size_t r = o; o += (bytes + 255) & ~(size_t)255; return r; };

    _Float16* x16   = (_Float16*)(ws + take((size_t)BN_ * 64 * 2));
    _Float16* xu16  = (_Float16*)(ws + take((size_t)BN_ * DXUP_ * 2));
    _Float16* k16   = (_Float16*)(ws + take((size_t)BN_ * 64 * 2));
    _Float16* v16t  = (_Float16*)(ws + take((size_t)B_ * 64 * N_ * 2));
    float*    wqf   = (float*)   (ws + take((size_t)BN_ * 4));
    float*    denom = (float*)   (ws + take(B_ * 4));
    float*    cvec  = (float*)   (ws + take(B_ * 64 * 4));
    _Float16* q16   = (_Float16*)(ws + take((size_t)B_ * P_ * 64 * 2));
    float*    pooled= (float*)   (ws + take((size_t)B_ * P_ * 64 * 4));
    _Float16* p16   = (_Float16*)(ws + take((size_t)B_ * P_ * 64 * 2));
    _Float16* kW1T  = (_Float16*)(ws + take(256 * 64 * 2));
    _Float16* kW2T  = (_Float16*)(ws + take(256 * 256 * 2));
    _Float16* kW3T  = (_Float16*)(ws + take(64 * 256 * 2));
    _Float16* vW1T  = (_Float16*)(ws + take(256 * DXUP_ * 2));
    _Float16* vW2T  = (_Float16*)(ws + take(256 * 256 * 2));
    _Float16* vW3T  = (_Float16*)(ws + take(64 * 256 * 2));
    _Float16* rW1T  = (_Float16*)(ws + take(256 * 64 * 2));
    _Float16* rW2T  = (_Float16*)(ws + take(128 * 256 * 2));

    const float* x_enc = (const float*)d_in[0];
    const float* u     = (const float*)d_in[1];
    const int*   mask  = (const int*)d_in[2];
    const float* sw    = (const float*)d_in[3];
    const float* kW1 = (const float*)d_in[4],  * kb1 = (const float*)d_in[5];
    const float* kW2 = (const float*)d_in[6],  * kb2 = (const float*)d_in[7];
    const float* kW3 = (const float*)d_in[8],  * kb3 = (const float*)d_in[9];
    const float* vW1 = (const float*)d_in[10], * vb1 = (const float*)d_in[11];
    const float* vW2 = (const float*)d_in[12], * vb2 = (const float*)d_in[13];
    const float* vW3 = (const float*)d_in[14], * vb3 = (const float*)d_in[15];
    const float* query = (const float*)d_in[16];
    const float* Bdirs = (const float*)d_in[17];
    const float* aW1 = (const float*)d_in[18], * ab1 = (const float*)d_in[19];
    const float* aW2 = (const float*)d_in[20], * ab2 = (const float*)d_in[21];
    const float* rW1 = (const float*)d_in[22], * rb1 = (const float*)d_in[23];
    const float* rW2 = (const float*)d_in[24], * rb2 = (const float*)d_in[25];

    // --- conversions ---
    conv_weight_kernel<<<(256 * 64 + 255) / 256, 256, 0, stream>>>(kW1, kW1T, 64, 64, 256);
    conv_weight_kernel<<<(256 * 256 + 255) / 256, 256, 0, stream>>>(kW2, kW2T, 256, 256, 256);
    conv_weight_kernel<<<(64 * 256 + 255) / 256, 256, 0, stream>>>(kW3, kW3T, 256, 256, 64);
    conv_weight_kernel<<<(256 * DXUP_ + 255) / 256, 256, 0, stream>>>(vW1, vW1T, 68, DXUP_, 256);
    conv_weight_kernel<<<(256 * 256 + 255) / 256, 256, 0, stream>>>(vW2, vW2T, 256, 256, 256);
    conv_weight_kernel<<<(64 * 256 + 255) / 256, 256, 0, stream>>>(vW3, vW3T, 256, 256, 64);
    conv_weight_kernel<<<(256 * 64 + 255) / 256, 256, 0, stream>>>(rW1, rW1T, 64, 64, 256);
    conv_weight_kernel<<<(128 * 256 + 255) / 256, 256, 0, stream>>>(rW2, rW2T, 256, 256, 128);
    conv_x16_kernel<<<BN_ * 64 / 256, 256, 0, stream>>>(x_enc, x16);
    conv_xu_kernel<<<BN_ * DXUP_ / 256, 256, 0, stream>>>(x_enc, u, xu16);
    conv_w_kernel<<<BN_ / 256, 256, 0, stream>>>(mask, sw, wqf);
    zero_kernel<<<B_ * P_ * 64 / 256, 256, 0, stream>>>(pooled);

    // --- key / value MLPs (WMMA), 4 waves/block, 1 wave = 16 rows ---
    mlp3_kernel<<<BN_ / 64, 128, 0, stream>>>(x16, 64, 2, kW1T, kb1, kW2T, kb2, kW3T, kb3, k16, 0);
    mlp3_kernel<<<BN_ / 64, 128, 0, stream>>>(xu16, DXUP_, 3, vW1T, vb1, vW2T, vb2, vW3T, vb3, v16t, 1);

    // --- context + adaptive queries ---
    ctx_kernel<<<B_, 256, 0, stream>>>(wqf, v16t, denom, cvec);
    adapt_kernel<<<B_, 128, 0, stream>>>(cvec, aW1, ab1, aW2, ab2, query, Bdirs, q16);

    // --- fused attention + pooling (WMMA) ---
    attn_kernel<<<B_ * 8 * NSPLIT_ / 4, 128, 0, stream>>>(q16, k16, v16t, wqf, pooled);
    pooled_norm_kernel<<<B_ * P_ * 64 / 256, 256, 0, stream>>>(pooled, denom, p16);

    // --- rho MLP (WMMA) -> f32 output ---
    mlp_rho_kernel<<<B_ * P_ / 64, 128, 0, stream>>>(p16, rW1T, rb1, rW2T, rb2, (float*)d_out);
}
